// CLIPEncoderLayer_70420283785327
// MI455X (gfx1250) — matrix operand order repaired and checked by reading, not compiled, not run on hardware
//
#include <hip/hip_runtime.h>
#include <hip/hip_bf16.h>

typedef __bf16 bf16;
typedef __attribute__((ext_vector_type(16))) __bf16 v16bf;
typedef __attribute__((ext_vector_type(8)))  __bf16 v8bf;
typedef __attribute__((ext_vector_type(8)))  float  v8f;
typedef __attribute__((ext_vector_type(4)))  unsigned int u32x4;
typedef __attribute__((ext_vector_type(8)))  int i32x8;
typedef __attribute__((ext_vector_type(4)))  int i32x4;

#define HDIM   768
#define FFDIM  3072
#define NHEADS 12
#define HEADD  64
#define SEQ    2048
#define BATCH  4

#if defined(__HIP_DEVICE_COMPILE__) && __has_builtin(__builtin_amdgcn_tensor_load_to_lds)
#define USE_TDM 1
#else
#define USE_TDM 0
#endif

// ---------------------------------------------------------------------------
// fragment helpers (CDNA5 wave32 WMMA VGPR layouts, ISA 7.12.2)
// ---------------------------------------------------------------------------
__device__ __forceinline__ v16bf cat8(v8bf lo, v8bf hi) {
  return __builtin_shufflevector(lo, hi, 0,1,2,3,4,5,6,7,8,9,10,11,12,13,14,15);
}
// A-matrix fragment: 8 bf16 at p, 8 bf16 at p+16  (k = [base..base+7] and [base+16..base+23])
__device__ __forceinline__ v16bf ldfrag_gap16(const bf16* p) {
  return cat8(*(const v8bf*)p, *(const v8bf*)(p + 16));
}
// 16 contiguous bf16 (k = base..base+15); unused, B fragments here use ldfrag_gap16
__device__ __forceinline__ v16bf ldfrag_c16(const bf16* p) {
  return cat8(*(const v8bf*)p, *(const v8bf*)(p + 8));
}

#if USE_TDM
// ---------------------------------------------------------------------------
// Tensor Data Mover: 2-D tile global->LDS (ISA 8.3/8.4 D# bitfields).
// data_size=2B. Optional LDS padding: 4 DWORDs (16B) after every 16 DWORDs
// (64B) -> matches a 32-elem bf16 tile row staged at stride 40 elems.
// clang-23 toolchain: 6-arg builtin (g0, g1, g2, g3, extra, cpol); groups 2/3
// and the extra operand are only used for >2-D / gather descriptors -> zeros.
// ---------------------------------------------------------------------------
__device__ __forceinline__ void tdm_load_2d(const void* gptr, unsigned lds_off,
                                            unsigned tensor_d0, unsigned tensor_d1,
                                            unsigned tile_d0, unsigned tile_d1,
                                            unsigned stride_elems, bool pad) {
  unsigned long long ga = (unsigned long long)(uintptr_t)gptr;
  u32x4 g0;
  g0[0] = 1u;                                          // count=1, user mode
  g0[1] = lds_off;                                     // lds_addr (bytes)
  g0[2] = (unsigned)(ga & 0xffffffffu);                // global_addr[31:0]
  g0[3] = (unsigned)((ga >> 32) & 0x01ffffffu) | (2u << 30);  // [56:32] | type=2
  unsigned d0 = (1u << 16);                            // data_size = 2 bytes
  if (pad) d0 |= (1u << 20) | (3u << 22) | (3u << 25); // pad_en, every 16 dw, +4 dw
  i32x8 g1;
  g1[0] = (int)d0;
  g1[1] = (int)((tensor_d0 & 0xffffu) << 16);          // tensor_dim0[15:0] @ bits 63:48
  g1[2] = (int)(((tensor_d0 >> 16) & 0xffffu) | ((tensor_d1 & 0xffffu) << 16));
  g1[3] = (int)(((tensor_d1 >> 16) & 0xffffu) | (tile_d0 << 16));  // tile_dim0 @127:112
  g1[4] = (int)(tile_d1 & 0xffffu);                    // tile_dim1 (tile_dim2 = 0)
  g1[5] = (int)stride_elems;                           // tensor_dim0_stride[31:0]
  g1[6] = 0;
  g1[7] = 0;
  i32x4 z4 = {0, 0, 0, 0};                             // 2-D tensor: groups 2/3 unused
  i32x8 z8 = {0, 0, 0, 0, 0, 0, 0, 0};
  __builtin_amdgcn_tensor_load_to_lds(g0, g1, z4, z4, z8, 0);
}
__device__ __forceinline__ unsigned lds_off_of(const void* p) {
  return (unsigned)(unsigned long long)(uintptr_t)p;   // generic LDS ptr low dword = offset
}
#endif

// ---------------------------------------------------------------------------
// weight fp32 [K,N] -> bf16 transposed [N,K]
// ---------------------------------------------------------------------------
__global__ void wtrans_cast(const float* __restrict__ W, bf16* __restrict__ Wt,
                            int K, int N) {
  int i = blockIdx.x * 256 + threadIdx.x;   // grid sized exactly K*N/256
  int k = i / N;
  int n = i - k * N;
  Wt[(size_t)n * K + k] = (bf16)W[i];
}

// ---------------------------------------------------------------------------
// LayerNorm (row = 768 fp32) -> bf16
// ---------------------------------------------------------------------------
__global__ __launch_bounds__(256) void layernorm_to_bf16(
    const float* __restrict__ X, const float* __restrict__ gamma,
    const float* __restrict__ beta, bf16* __restrict__ out) {
  __shared__ float red[256];
  const int row = blockIdx.x, tid = threadIdx.x;
  const float* xr = X + (size_t)row * HDIM;
  float v0 = xr[tid], v1 = xr[tid + 256], v2 = xr[tid + 512];
  red[tid] = v0 + v1 + v2;
  __syncthreads();
  for (int o = 128; o > 0; o >>= 1) { if (tid < o) red[tid] += red[tid + o]; __syncthreads(); }
  float mean = red[0] * (1.0f / HDIM);
  __syncthreads();
  float d0 = v0 - mean, d1 = v1 - mean, d2 = v2 - mean;
  red[tid] = d0 * d0 + d1 * d1 + d2 * d2;
  __syncthreads();
  for (int o = 128; o > 0; o >>= 1) { if (tid < o) red[tid] += red[tid + o]; __syncthreads(); }
  float rstd = rsqrtf(red[0] * (1.0f / HDIM) + 1e-5f);
  bf16* orow = out + (size_t)row * HDIM;
  orow[tid]       = (bf16)(d0 * rstd * gamma[tid]       + beta[tid]);
  orow[tid + 256] = (bf16)(d1 * rstd * gamma[tid + 256] + beta[tid + 256]);
  orow[tid + 512] = (bf16)(d2 * rstd * gamma[tid + 512] + beta[tid + 512]);
}

// ---------------------------------------------------------------------------
// bf16 WMMA GEMM: C[M,N] = epi(A[M,K] * Wt[N,K]^T + bias [+ resid])
// block tile 128x128x32, 8 waves (2M x 4N), 8 WMMAs per wave per k-step.
// Tiles staged to LDS by the Tensor Data Mover (wave 0 issues, TENSORcnt
// gated, workgroup barrier releases consumers).
// ---------------------------------------------------------------------------
constexpr int EPI_BF16      = 0;  // bf16 out, +bias
constexpr int EPI_F32_RES   = 1;  // fp32 out, +bias +resid(fp32)
constexpr int EPI_BF16_GELU = 2;  // bf16 out, +bias, exact gelu

template <int EPI>
__global__ __launch_bounds__(256) void gemm_bf16_wmma(
    const bf16* __restrict__ A,    // [M,K] row-major
    const bf16* __restrict__ Wt,   // [N,K] row-major (pre-transposed weight)
    const float* __restrict__ bias,
    const float* __restrict__ resid,
    void* __restrict__ Cout, int M, int N, int K) {
  constexpr int BM = 128, BN = 128, BK = 32, LDT = BK + 8;
  __shared__ bf16 As[BM * LDT];
  __shared__ bf16 Bs[BN * LDT];
  const int tid = threadIdx.x;
  const int wid = tid >> 5, lane = tid & 31, half = lane >> 4, l15 = lane & 15;
  const int waveM = wid & 1, waveN = wid >> 1;           // 2 x 4 wave grid
  const int rowBase = blockIdx.y * BM, colBase = blockIdx.x * BN;

  const v8f vzero = {0.f, 0.f, 0.f, 0.f, 0.f, 0.f, 0.f, 0.f};
  v8f acc[4][2];
#pragma unroll
  for (int mt = 0; mt < 4; mt++)
#pragma unroll
    for (int nt = 0; nt < 2; nt++) acc[mt][nt] = vzero;

#if USE_TDM
  const unsigned ldsA = lds_off_of(As), ldsB = lds_off_of(Bs);
#endif

  for (int kb = 0; kb < K; kb += BK) {
#if USE_TDM
    if (wid == 0) {
      tdm_load_2d(A  + (size_t)rowBase * K + kb, ldsA, (unsigned)K, (unsigned)M,
                  BK, BM, (unsigned)K, true);
      tdm_load_2d(Wt + (size_t)colBase * K + kb, ldsB, (unsigned)K, (unsigned)N,
                  BK, BN, (unsigned)K, true);
      __builtin_amdgcn_s_wait_tensorcnt(0);
    }
#else
    // manual staging: 512 x 16B chunks per tile, 2 per thread
#pragma unroll
    for (int i = 0; i < 2; i++) {
      int c = tid + i * 256;
      int r = c >> 2, off = (c & 3) * 8;
      *(uint4*)&As[r * LDT + off] =
          *(const uint4*)(A + (size_t)(rowBase + r) * K + kb + off);
      *(uint4*)&Bs[r * LDT + off] =
          *(const uint4*)(Wt + (size_t)(colBase + r) * K + kb + off);
    }
#endif
    if (kb + BK < K) {  // prefetch next tiles into cache (global_prefetch_b8)
      __builtin_prefetch(A + (size_t)(rowBase + (tid >> 1)) * K + kb + BK, 0, 1);
      __builtin_prefetch(Wt + (size_t)(colBase + (tid >> 1)) * K + kb + BK, 0, 1);
    }
    __syncthreads();

    v16bf af[4], bfm[2];
#pragma unroll
    for (int mt = 0; mt < 4; mt++) {
      int m = waveM * 64 + mt * 16 + l15;
      af[mt] = ldfrag_gap16(&As[m * LDT + half * 8]);
    }
#pragma unroll
    for (int nt = 0; nt < 2; nt++) {
      int n = waveN * 32 + nt * 16 + l15;
      bfm[nt] = ldfrag_gap16(&Bs[n * LDT + half * 8]);
    }
#pragma unroll
    for (int mt = 0; mt < 4; mt++)
#pragma unroll
      for (int nt = 0; nt < 2; nt++)
        acc[mt][nt] = __builtin_amdgcn_wmma_f32_16x16x32_bf16(
            false, af[mt], false, bfm[nt], (short)0, acc[mt][nt], false, false);
    __syncthreads();
  }

#pragma unroll
  for (int mt = 0; mt < 4; mt++) {
#pragma unroll
    for (int nt = 0; nt < 2; nt++) {
      int col = colBase + waveN * 32 + nt * 16 + l15;
      int row0 = rowBase + waveM * 64 + mt * 16 + half * 8;
      float bc = bias[col];
#pragma unroll
      for (int v = 0; v < 8; v++) {
        size_t idx = (size_t)(row0 + v) * N + col;
        float val = acc[mt][nt][v] + bc;
        if constexpr (EPI == EPI_F32_RES) {
          ((float*)Cout)[idx] = val + resid[idx];
        } else if constexpr (EPI == EPI_BF16_GELU) {
          val = 0.5f * val * (1.0f + erff(val * 0.70710678118654752f));
          ((bf16*)Cout)[idx] = (bf16)val;
        } else {
          ((bf16*)Cout)[idx] = (bf16)val;
        }
      }
    }
  }
}

// ---------------------------------------------------------------------------
// Flash attention, WMMA bf16. grid = (S/128, NH, B), 8 waves; each wave owns
// 16 query rows; streams key/value tiles of 64 through LDS. The K-tile is a
// straight row-major copy -> TDM; the V-tile needs a transpose -> VALU path
// overlapped with the TDM transfer.
// ---------------------------------------------------------------------------
__global__ __launch_bounds__(256) void flash_attn_wmma(
    const bf16* __restrict__ Q, const bf16* __restrict__ Kb,
    const bf16* __restrict__ Vb, const int* __restrict__ amask,
    bf16* __restrict__ ctx) {
  constexpr float SCALE = 0.125f;  // HD^-0.5
  __shared__ bf16 Ks[64 * 64];       // [key][d]
  __shared__ bf16 Vts[64 * 72];      // [d][key] (transposed, padded)
  __shared__ bf16 Ps[8 * 16 * 64];   // per-wave P relayout scratch
  __shared__ float msk[64];
  const int tid = threadIdx.x;
  const int wid = tid >> 5, lane = tid & 31, half = lane >> 4, l15 = lane & 15;
  const int bb = blockIdx.z, h = blockIdx.y;
  const int q0 = blockIdx.x * 128 + wid * 16;

  // Q A-fragments (persistent): rows m = l15, contraction over d (2 x K=32)
  const bf16* qp = Q + (size_t)(bb * SEQ + q0 + l15) * HDIM + h * HEADD;
  v16bf qa0 = ldfrag_gap16(qp + half * 8);
  v16bf qa1 = ldfrag_gap16(qp + 32 + half * 8);

  const v8f vzero = {0.f, 0.f, 0.f, 0.f, 0.f, 0.f, 0.f, 0.f};
  v8f mrun, lrun, Ot[4];
#pragma unroll
  for (int v = 0; v < 8; v++) { mrun[v] = -1e30f; lrun[v] = 0.f; }
#pragma unroll
  for (int t = 0; t < 4; t++) Ot[t] = vzero;

#if USE_TDM
  const unsigned ldsK = lds_off_of(Ks);
#endif

  for (int kt = 0; kt < SEQ; kt += 64) {
#if USE_TDM
    if (wid == 0)   // K-tile via Tensor Data Mover (row-major 64x64, no pad)
      tdm_load_2d(Kb + (size_t)(bb * SEQ + kt) * HDIM + h * HEADD, ldsK,
                  HDIM, BATCH * SEQ, 64, 64, HDIM, false);
#endif
#pragma unroll
    for (int i = 0; i < 2; i++) {
      int c = tid + i * 256;
      int r = c >> 3, off = (c & 7) * 8;
#if !USE_TDM
      *(uint4*)&Ks[r * 64 + off] =
          *(const uint4*)(Kb + (size_t)(bb * SEQ + kt + r) * HDIM + h * HEADD + off);
#endif
      v8bf vv = *(const v8bf*)(Vb + (size_t)(bb * SEQ + kt + r) * HDIM + h * HEADD + off);
#pragma unroll
      for (int j = 0; j < 8; j++) Vts[(off + j) * 72 + r] = vv[j];  // transpose
    }
    if (tid < 64) msk[tid] = (amask[bb * SEQ + kt + tid] != 0) ? 0.f : -1e30f;
#if USE_TDM
    if (wid == 0) __builtin_amdgcn_s_wait_tensorcnt(0);
#endif
    __syncthreads();

    // scores S = Q K^T  (contraction over d, B-frag reads Ks[key][d] contiguously)
    v8f sc[4];
#pragma unroll
    for (int nt = 0; nt < 4; nt++) {
      v8f a = vzero;
      v16bf b0 = ldfrag_gap16(&Ks[(nt * 16 + l15) * 64 + half * 8]);
      a = __builtin_amdgcn_wmma_f32_16x16x32_bf16(false, qa0, false, b0, (short)0, a, false, false);
      v16bf b1 = ldfrag_gap16(&Ks[(nt * 16 + l15) * 64 + 32 + half * 8]);
      a = __builtin_amdgcn_wmma_f32_16x16x32_bf16(false, qa1, false, b1, (short)0, a, false, false);
      float mk = msk[nt * 16 + l15];
#pragma unroll
      for (int v = 0; v < 8; v++) sc[nt][v] = a[v] * SCALE + mk;
    }

    // streaming softmax (row = v + 8*half; cols live across 16-lane group)
    v8f rmax;
#pragma unroll
    for (int v = 0; v < 8; v++)
      rmax[v] = fmaxf(fmaxf(sc[0][v], sc[1][v]), fmaxf(sc[2][v], sc[3][v]));
#pragma unroll
    for (int d = 1; d < 16; d <<= 1)
#pragma unroll
      for (int v = 0; v < 8; v++) rmax[v] = fmaxf(rmax[v], __shfl_xor(rmax[v], d, 32));
    v8f mnew, alpha, rsum = vzero;
#pragma unroll
    for (int v = 0; v < 8; v++) {
      mnew[v] = fmaxf(mrun[v], rmax[v]);
      alpha[v] = expf(mrun[v] - mnew[v]);
    }
#pragma unroll
    for (int nt = 0; nt < 4; nt++)
#pragma unroll
      for (int v = 0; v < 8; v++) {
        float p = expf(sc[nt][v] - mnew[v]);
        sc[nt][v] = p;
        rsum[v] += p;
      }
#pragma unroll
    for (int d = 1; d < 16; d <<= 1)
#pragma unroll
      for (int v = 0; v < 8; v++) rsum[v] += __shfl_xor(rsum[v], d, 32);
#pragma unroll
    for (int v = 0; v < 8; v++) {
      lrun[v] = lrun[v] * alpha[v] + rsum[v];
      mrun[v] = mnew[v];
    }
#pragma unroll
    for (int t = 0; t < 4; t++)
#pragma unroll
      for (int v = 0; v < 8; v++) Ot[t][v] *= alpha[v];

    // relayout P (D layout -> A layout) through per-wave LDS; same-wave DS ops
    // are in-order on CDNA5, no barrier needed.
    bf16* pw = &Ps[wid * 16 * 64];
#pragma unroll
    for (int nt = 0; nt < 4; nt++)
#pragma unroll
      for (int v = 0; v < 8; v++)
        pw[(v + 8 * half) * 64 + nt * 16 + l15] = (bf16)sc[nt][v];

    // O += P V   (contraction over keys; B-frag reads Vts[d][key] contiguously)
#pragma unroll
    for (int kc = 0; kc < 2; kc++) {
      v16bf af = ldfrag_gap16(&pw[l15 * 64 + kc * 32 + half * 8]);
#pragma unroll
      for (int nt = 0; nt < 4; nt++) {
        v16bf bfv = ldfrag_gap16(&Vts[(nt * 16 + l15) * 72 + kc * 32 + half * 8]);
        Ot[nt] = __builtin_amdgcn_wmma_f32_16x16x32_bf16(
            false, af, false, bfv, (short)0, Ot[nt], false, false);
      }
    }
    __syncthreads();
  }

#pragma unroll
  for (int nt = 0; nt < 4; nt++)
#pragma unroll
    for (int v = 0; v < 8; v++) {
      float val = Ot[nt][v] / lrun[v];
      size_t row = (size_t)(bb * SEQ + q0 + v + 8 * half);
      ctx[row * HDIM + h * HEADD + nt * 16 + l15] = (bf16)val;
    }
}

// ---------------------------------------------------------------------------
// launch
// ---------------------------------------------------------------------------
extern "C" void kernel_launch(void* const* d_in, const int* in_sizes, int n_in,
                              void* d_out, int out_size, void* d_ws, size_t ws_size,
                              hipStream_t stream) {
  (void)in_sizes; (void)n_in; (void)out_size; (void)ws_size;
  const float* hidden = (const float*)d_in[0];
  const int*   amask  = (const int*)d_in[1];
  const float* ln1_g  = (const float*)d_in[2];
  const float* ln1_b  = (const float*)d_in[3];
  const float* wq = (const float*)d_in[4];   const float* bq = (const float*)d_in[5];
  const float* wk = (const float*)d_in[6];   const float* bk = (const float*)d_in[7];
  const float* wv = (const float*)d_in[8];   const float* bv = (const float*)d_in[9];
  const float* wo = (const float*)d_in[10];  const float* bo = (const float*)d_in[11];
  const float* ln2_g = (const float*)d_in[12];
  const float* ln2_b = (const float*)d_in[13];
  const float* w1 = (const float*)d_in[14];  const float* b1 = (const float*)d_in[15];
  const float* w2 = (const float*)d_in[16];  const float* b2 = (const float*)d_in[17];

  const int M = BATCH * SEQ;  // 8192
  char* ws = (char*)d_ws;
  size_t off = 0;
  auto take = [&](size_t bytes) -> char* {
    char* p = ws + off;
    off += (bytes + 255) & ~(size_t)255;
    return p;
  };
  bf16* wqT = (bf16*)take((size_t)HDIM * HDIM * 2);
  bf16* wkT = (bf16*)take((size_t)HDIM * HDIM * 2);
  bf16* wvT = (bf16*)take((size_t)HDIM * HDIM * 2);
  bf16* woT = (bf16*)take((size_t)HDIM * HDIM * 2);
  bf16* w1T = (bf16*)take((size_t)HDIM * FFDIM * 2);
  bf16* w2T = (bf16*)take((size_t)HDIM * FFDIM * 2);
  bf16* xln = (bf16*)take((size_t)M * HDIM * 2);
  bf16* qb  = (bf16*)take((size_t)M * HDIM * 2);
  bf16* kb  = (bf16*)take((size_t)M * HDIM * 2);
  bf16* vb  = (bf16*)take((size_t)M * HDIM * 2);
  bf16* ctx = (bf16*)take((size_t)M * HDIM * 2);
  float* x1 = (float*)take((size_t)M * HDIM * 4);
  bf16* yln = (bf16*)take((size_t)M * HDIM * 2);
  bf16* hff = (bf16*)take((size_t)M * FFDIM * 2);

  // weights -> bf16, transposed [N][K]
  wtrans_cast<<<HDIM * HDIM / 256, 256, 0, stream>>>(wq, wqT, HDIM, HDIM);
  wtrans_cast<<<HDIM * HDIM / 256, 256, 0, stream>>>(wk, wkT, HDIM, HDIM);
  wtrans_cast<<<HDIM * HDIM / 256, 256, 0, stream>>>(wv, wvT, HDIM, HDIM);
  wtrans_cast<<<HDIM * HDIM / 256, 256, 0, stream>>>(wo, woT, HDIM, HDIM);
  wtrans_cast<<<HDIM * FFDIM / 256, 256, 0, stream>>>(w1, w1T, HDIM, FFDIM);
  wtrans_cast<<<HDIM * FFDIM / 256, 256, 0, stream>>>(w2, w2T, FFDIM, HDIM);

  // LN1
  layernorm_to_bf16<<<M, 256, 0, stream>>>(hidden, ln1_g, ln1_b, xln);

  // QKV projections
  dim3 gH(HDIM / 128, M / 128);
  gemm_bf16_wmma<EPI_BF16><<<gH, 256, 0, stream>>>(xln, wqT, bq, nullptr, qb, M, HDIM, HDIM);
  gemm_bf16_wmma<EPI_BF16><<<gH, 256, 0, stream>>>(xln, wkT, bk, nullptr, kb, M, HDIM, HDIM);
  gemm_bf16_wmma<EPI_BF16><<<gH, 256, 0, stream>>>(xln, wvT, bv, nullptr, vb, M, HDIM, HDIM);

  // attention
  dim3 ga(SEQ / 128, NHEADS, BATCH);
  flash_attn_wmma<<<ga, 256, 0, stream>>>(qb, kb, vb, amask, ctx);

  // output projection + residual (fp32)
  gemm_bf16_wmma<EPI_F32_RES><<<gH, 256, 0, stream>>>(ctx, woT, bo, hidden, x1, M, HDIM, HDIM);

  // LN2 + MLP
  layernorm_to_bf16<<<M, 256, 0, stream>>>(x1, ln2_g, ln2_b, yln);
  dim3 gF(FFDIM / 128, M / 128);
  gemm_bf16_wmma<EPI_BF16_GELU><<<gF, 256, 0, stream>>>(yln, w1T, b1, nullptr, hff, M, FFDIM, HDIM);
  gemm_bf16_wmma<EPI_F32_RES><<<gH, 256, 0, stream>>>(hff, w2T, b2, x1, (float*)d_out, M, HDIM, FFDIM);
}
